// MentionScorer_40011915329911
// MI455X (gfx1250) — matrix operand order repaired and checked by reading, not compiled, not run on hardware
//
#include <hip/hip_runtime.h>

#define T_TOK   4096
#define STATE_D 1024
#define EMBED_D 512
#define DIST_D  20
#define HID     1024
#define GI_D    2580   // 2*STATE_D + EMBED_D + DIST_D
#define GI_PAD  2624   // GI_D rounded up to 64 (KT)
#define MAXW    10

typedef __attribute__((ext_vector_type(16))) __bf16         v16bf;
typedef __attribute__((ext_vector_type(8)))  float          v8f;
typedef __attribute__((ext_vector_type(8)))  unsigned short ushort8;
typedef __attribute__((ext_vector_type(4)))  unsigned short usx4;   // avoid HIP's ushort4

__device__ __forceinline__ unsigned short f2bf(float f) {
    unsigned int u = __float_as_uint(f);
    return (unsigned short)((u + 0x7FFFu + ((u >> 16) & 1u)) >> 16);  // RNE
}
__device__ __forceinline__ float bf2f(unsigned short h) {
    return __uint_as_float(((unsigned int)h) << 16);
}
__device__ __forceinline__ usx4 pack_bf16x4(float4 a) {
    usx4 r;
    r[0] = f2bf(a.x); r[1] = f2bf(a.y); r[2] = f2bf(a.z); r[3] = f2bf(a.w);
    return r;
}
// CDNA5 async global->LDS 16B copy (per-lane scatter into padded LDS layout).
__device__ __forceinline__ void async_copy_b128(unsigned lds_byte_off, const void* gaddr) {
    asm volatile("global_load_async_to_lds_b128 %0, %1, off"
                 :: "v"(lds_byte_off), "v"(gaddr) : "memory");
}
__device__ __forceinline__ void wait_async0() {
    asm volatile("s_wait_asynccnt 0x0" ::: "memory");
}
__device__ __forceinline__ void wait_async8() {
    asm volatile("s_wait_asynccnt 0x8" ::: "memory");
}
__device__ __forceinline__ unsigned lds_off_of(const void* p) {
    return (unsigned)(uintptr_t)p;  // low 32 bits of flat shared address = LDS offset
}

// ------------------------------------------------------------------
// weight prep: f32 [K,N] -> bf16 transposed [N,Kpad], zero-padded tail
// grid: x covers N (coalesced reads), y = Kpad
// ------------------------------------------------------------------
__global__ void k_w_prep(const float* __restrict__ in, unsigned short* __restrict__ out,
                         int K, int N, int Kpad) {
    int n = blockIdx.x * blockDim.x + threadIdx.x;
    int k = blockIdx.y;
    if (n < N)
        out[(size_t)n * Kpad + k] = (k < K) ? f2bf(in[(size_t)k * N + n]) : (unsigned short)0;
}

// ------------------------------------------------------------------
// bulk f32 -> bf16 (contiguous, count multiple of 8)
// ------------------------------------------------------------------
__global__ void k_cvt_bf16(const float* __restrict__ in, unsigned short* __restrict__ out,
                           size_t n8) {   // n8 = count/8
    for (size_t i = blockIdx.x * (size_t)blockDim.x + threadIdx.x; i < n8;
         i += (size_t)gridDim.x * blockDim.x) {
        float4 a = ((const float4*)in)[i * 2];
        float4 b = ((const float4*)in)[i * 2 + 1];
        ushort8 r;
        usx4 lo = pack_bf16x4(a), hi = pack_bf16x4(b);
        r[0]=lo[0]; r[1]=lo[1]; r[2]=lo[2]; r[3]=lo[3];
        r[4]=hi[0]; r[5]=hi[1]; r[6]=hi[2]; r[7]=hi[3];
        ((ushort8*)out)[i] = r;
    }
}

// ------------------------------------------------------------------
// Tiled WMMA GEMM:  out[M,N](bf16) = relu( A[M,Kpad](bf16) @ W + bias )
// W pre-transposed bf16: Wt[N][Kpad]. Kpad % 64 == 0, zero-padded.
// BM=128, BN=128, KT=64. 256 threads = 8 waves (4 M x 2 N); wave tile
// 32x64 -> 16 v_wmma_f32_16x16x32_bf16 per k-tile. Double-buffered LDS
// fed by global_load_async_to_lds_b128 (8 asyncs/thread/tile), fenced
// with s_wait_asynccnt; last iteration peeled so hot loop is branch-free.
// ------------------------------------------------------------------
__global__ __launch_bounds__(256) void k_gemm_bf16(
    const unsigned short* __restrict__ A, const unsigned short* __restrict__ Wt,
    const float* __restrict__ bias, unsigned short* __restrict__ out,
    int M, int N, int Kpad, int do_relu) {
    constexpr int BM = 128, BN = 128, KT = 64;
    constexpr int ASTR = 72, BSTR = 72;            // padded LDS strides (bank spread)
    __shared__ unsigned short As[2][BM * ASTR];    // 2 x 18 KB
    __shared__ unsigned short Bs[2][BN * BSTR];    // 2 x 18 KB

    const int tid   = threadIdx.x;
    const int wave  = tid >> 5, lane = tid & 31;
    const int waveM = wave >> 1, waveN = wave & 1;
    const int m0 = blockIdx.y * BM, n0 = blockIdx.x * BN;
    const int mrow = lane & 15, khalf = lane >> 4; // ISA 16-bit WMMA lane split

    // staging coords: 2 threads per 64-elem row, 4x16B chunks each
    const int srow = tid >> 1, shalf = tid & 1;
    int gmc = m0 + srow; if (gmc >= M) gmc = M - 1;         // clamp; masked in epilogue

    auto stage = [&](int k0, int buf) {
        const unsigned short* asrc = A + (size_t)gmc * Kpad + k0 + shalf * 32;
        unsigned adst = lds_off_of(&As[buf][srow * ASTR + shalf * 32]);
        async_copy_b128(adst,      asrc);
        async_copy_b128(adst + 16, asrc + 8);
        async_copy_b128(adst + 32, asrc + 16);
        async_copy_b128(adst + 48, asrc + 24);
        const unsigned short* bsrc = Wt + (size_t)(n0 + srow) * Kpad + k0 + shalf * 32;
        unsigned bdst = lds_off_of(&Bs[buf][srow * BSTR + shalf * 32]);
        async_copy_b128(bdst,      bsrc);
        async_copy_b128(bdst + 16, bsrc + 8);
        async_copy_b128(bdst + 32, bsrc + 16);
        async_copy_b128(bdst + 48, bsrc + 24);
    };

    v8f acc[2][4] = {};

    auto compute = [&](int buf) {
        union FragCvt { ushort8 h[2]; v16bf v; };
        #pragma unroll
        for (int kk = 0; kk < 2; ++kk) {          // two 16x16x32 k-steps per tile
            v16bf af[2], bfr[4];
            #pragma unroll
            for (int i = 0; i < 2; i++) {
                const unsigned short* p =
                    &As[buf][(waveM * 32 + i * 16 + mrow) * ASTR + kk * 32];
                FragCvt fc;
                fc.h[0] = *(const ushort8*)(p + 8 * khalf);       // K = 8*khalf..
                fc.h[1] = *(const ushort8*)(p + 16 + 8 * khalf);  // K = 16+8*khalf..
                af[i] = fc.v;
            }
            #pragma unroll
            for (int j = 0; j < 4; j++) {
                const unsigned short* p =
                    &Bs[buf][(waveN * 64 + j * 16 + mrow) * BSTR + kk * 32];
                FragCvt fc;
                fc.h[0] = *(const ushort8*)(p + 8 * khalf);       // K = 8*khalf..
                fc.h[1] = *(const ushort8*)(p + 16 + 8 * khalf);  // K = 16+8*khalf..
                bfr[j] = fc.v;
            }
            #pragma unroll
            for (int i = 0; i < 2; i++)
                #pragma unroll
                for (int j = 0; j < 4; j++)
                    acc[i][j] = __builtin_amdgcn_wmma_f32_16x16x32_bf16(
                        false, af[i], false, bfr[j], (short)0, acc[i][j], false, false);
        }
    };

    const int nt = Kpad / KT;                     // >= 2 for all our shapes
    stage(0, 0);
    int t = 0;
    for (; t < nt - 1; ++t) {                     // branch-free steady state
        stage((t + 1) * KT, (t & 1) ^ 1);
        wait_async8();                            // oldest 8 (tile t) landed
        __syncthreads();
        compute(t & 1);
        __syncthreads();
    }
    wait_async0();                                // final tile
    __syncthreads();
    compute(t & 1);

    // epilogue: C/D layout -> row = r + 8*khalf, col = lane%16
    #pragma unroll
    for (int i = 0; i < 2; i++) {
        #pragma unroll
        for (int j = 0; j < 4; j++) {
            int gn = n0 + waveN * 64 + j * 16 + mrow;
            float bv = bias[gn];
            #pragma unroll
            for (int r = 0; r < 8; r++) {
                int gm = m0 + waveM * 32 + i * 16 + khalf * 8 + r;
                if (gm < M) {
                    float v = acc[i][j][r] + bv;
                    if (do_relu) v = v > 0.f ? v : 0.f;
                    out[(size_t)gm * N + gn] = f2bf(v);
                }
            }
        }
    }
}

// ------------------------------------------------------------------
// final layer matvec: out[m] = sum_k bf2f(h[m,k]) * w[k] + b[0]
// one wave32 per row, 16B vector loads
// ------------------------------------------------------------------
__global__ __launch_bounds__(256) void k_matvec(
    const unsigned short* __restrict__ h, const float* __restrict__ w,
    const float* __restrict__ b, float* __restrict__ out, int M, int K) {
    int row  = blockIdx.x * 8 + (threadIdx.x >> 5);
    int lane = threadIdx.x & 31;
    if (row >= M) return;
    const unsigned short* hp = h + (size_t)row * K;
    float acc = 0.f;
    for (int base = lane * 8; base < K; base += 32 * 8) {
        ushort8 v = *(const ushort8*)(hp + base);
        float4 w0 = ((const float4*)(w + base))[0];
        float4 w1 = ((const float4*)(w + base))[1];
        acc += bf2f(v[0]) * w0.x + bf2f(v[1]) * w0.y + bf2f(v[2]) * w0.z + bf2f(v[3]) * w0.w;
        acc += bf2f(v[4]) * w1.x + bf2f(v[5]) * w1.y + bf2f(v[6]) * w1.z + bf2f(v[7]) * w1.w;
    }
    #pragma unroll
    for (int off = 16; off > 0; off >>= 1) acc += __shfl_down(acc, off, 32);
    if (lane == 0) out[row] = acc + b[0];
}

// ------------------------------------------------------------------
// span pooling: builds g_i rows -> f32 [N,2580] in d_out AND bf16
// zero-padded mirror [N,GI_PAD] in workspace (GEMM input).
// ------------------------------------------------------------------
__global__ __launch_bounds__(128) void k_span_pool(
    const float* __restrict__ states, const float* __restrict__ embeds,
    const float* __restrict__ attns, const float* __restrict__ width_embed,
    const int* __restrict__ starts, const int* __restrict__ widths,
    float* __restrict__ g_out, unsigned short* __restrict__ g_bf, int N) {
    int n = blockIdx.x;
    if (n >= N) return;
    int s = starts[n], w = widths[n];

    float wt[MAXW];
    float mx = -3.4e38f;
    #pragma unroll
    for (int i = 0; i < MAXW; i++) {
        float a = (i < w) ? attns[s + i] : -1e10f;
        wt[i] = a;
        mx = fmaxf(mx, a);
    }
    float sum = 0.f;
    #pragma unroll
    for (int i = 0; i < MAXW; i++) { wt[i] = __expf(wt[i] - mx); sum += wt[i]; }
    float inv = 1.f / sum;
    #pragma unroll
    for (int i = 0; i < MAXW; i++) wt[i] *= inv;

    int e = s + w - 1;
    float4* go4 = (float4*)(g_out + (size_t)n * GI_D);
    usx4*   gb4 = (usx4*)(g_bf + (size_t)n * GI_PAD);
    const float4* rs4 = (const float4*)(states + (size_t)s * STATE_D);
    const float4* re4 = (const float4*)(states + (size_t)e * STATE_D);
    for (int d = threadIdx.x; d < STATE_D / 4; d += 128) {   // 256 float4s
        float4 v0 = rs4[d], v1 = re4[d];
        go4[d]               = v0;
        go4[STATE_D / 4 + d] = v1;
        gb4[d]               = pack_bf16x4(v0);
        gb4[STATE_D / 4 + d] = pack_bf16x4(v1);
    }
    for (int d = threadIdx.x; d < EMBED_D / 4; d += 128) {   // 128 float4s
        float4 acc = make_float4(0.f, 0.f, 0.f, 0.f);
        #pragma unroll
        for (int i = 0; i < MAXW; i++) {
            int p = s + i;
            p = p < (T_TOK - 1) ? p : (T_TOK - 1);
            float4 ev = ((const float4*)(embeds + (size_t)p * EMBED_D))[d];
            acc.x += wt[i] * ev.x; acc.y += wt[i] * ev.y;
            acc.z += wt[i] * ev.z; acc.w += wt[i] * ev.w;
        }
        go4[2 * STATE_D / 4 + d] = acc;
        gb4[2 * STATE_D / 4 + d] = pack_bf16x4(acc);
    }
    const int bins[8] = {1, 2, 3, 4, 8, 16, 32, 64};
    int bucket = 0;
    #pragma unroll
    for (int i = 0; i < 8; i++) bucket += (w >= bins[i]) ? 1 : 0;
    const float4* we4 = (const float4*)(width_embed + (size_t)bucket * DIST_D);
    for (int d = threadIdx.x; d < DIST_D / 4; d += 128) {    // 5 float4s
        float4 v = we4[d];
        go4[(2 * STATE_D + EMBED_D) / 4 + d] = v;
        gb4[(2 * STATE_D + EMBED_D) / 4 + d] = pack_bf16x4(v);
    }
    // zero the 44-element K pad (2580..2623)
    usx4 z = {0, 0, 0, 0};
    for (int d = threadIdx.x; d < (GI_PAD - GI_D) / 4; d += 128)  // 11 usx4s
        gb4[GI_D / 4 + d] = z;
}

// ------------------------------------------------------------------
extern "C" void kernel_launch(void* const* d_in, const int* in_sizes, int n_in,
                              void* d_out, int out_size, void* d_ws, size_t ws_size,
                              hipStream_t stream) {
    (void)n_in; (void)out_size; (void)ws_size;
    const float* states      = (const float*)d_in[0];
    const float* embeds      = (const float*)d_in[1];
    const float* attn_W1     = (const float*)d_in[2];
    const float* attn_b1     = (const float*)d_in[3];
    const float* attn_W2     = (const float*)d_in[4];
    const float* attn_b2     = (const float*)d_in[5];
    const float* attn_W3     = (const float*)d_in[6];
    const float* attn_b3     = (const float*)d_in[7];
    const float* score_W1    = (const float*)d_in[8];
    const float* score_b1    = (const float*)d_in[9];
    const float* score_W2    = (const float*)d_in[10];
    const float* score_b2    = (const float*)d_in[11];
    const float* score_W3    = (const float*)d_in[12];
    const float* score_b3    = (const float*)d_in[13];
    const float* width_embed = (const float*)d_in[14];
    const int*   starts      = (const int*)d_in[15];
    const int*   widths      = (const int*)d_in[16];
    const int N = in_sizes[15];   // 40915 spans

    // workspace carve-up (~420 MB)
    char* ws = (char*)d_ws;
    size_t off = 0;
    auto alloc = [&](size_t bytes) -> char* {
        char* p = ws + off;
        off += (bytes + 255) & ~(size_t)255;
        return p;
    };
    unsigned short* wA1t = (unsigned short*)alloc((size_t)HID * STATE_D * 2);
    unsigned short* wA2t = (unsigned short*)alloc((size_t)HID * HID * 2);
    unsigned short* wS1t = (unsigned short*)alloc((size_t)HID * GI_PAD * 2);
    unsigned short* wS2t = (unsigned short*)alloc((size_t)HID * HID * 2);
    unsigned short* stbf = (unsigned short*)alloc((size_t)T_TOK * STATE_D * 2);
    unsigned short* hA1  = (unsigned short*)alloc((size_t)T_TOK * HID * 2);
    unsigned short* hA2  = (unsigned short*)alloc((size_t)T_TOK * HID * 2);
    float*          atn  = (float*)alloc((size_t)T_TOK * 4);
    unsigned short* gbf  = (unsigned short*)alloc((size_t)N * GI_PAD * 2);
    unsigned short* hS1  = (unsigned short*)alloc((size_t)N * HID * 2);
    unsigned short* hS2  = (unsigned short*)alloc((size_t)N * HID * 2);

    // weights -> bf16 transposed [N][Kpad]; states -> bf16 (one-shot)
    k_w_prep<<<dim3((HID + 255) / 256, STATE_D), 256, 0, stream>>>(attn_W1,  wA1t, STATE_D, HID, STATE_D);
    k_w_prep<<<dim3((HID + 255) / 256, HID),     256, 0, stream>>>(attn_W2,  wA2t, HID,     HID, HID);
    k_w_prep<<<dim3((HID + 255) / 256, GI_PAD),  256, 0, stream>>>(score_W1, wS1t, GI_D,    HID, GI_PAD);
    k_w_prep<<<dim3((HID + 255) / 256, HID),     256, 0, stream>>>(score_W2, wS2t, HID,     HID, HID);
    k_cvt_bf16<<<2048, 256, 0, stream>>>(states, stbf, (size_t)T_TOK * STATE_D / 8);

    dim3 blk(256);
    // token attention MLP
    dim3 gA(HID / 128, (T_TOK + 127) / 128);
    k_gemm_bf16<<<gA, blk, 0, stream>>>(stbf, wA1t, attn_b1, hA1, T_TOK, HID, STATE_D, 1);
    k_gemm_bf16<<<gA, blk, 0, stream>>>(hA1,  wA2t, attn_b2, hA2, T_TOK, HID, HID, 1);
    k_matvec<<<(T_TOK + 7) / 8, 256, 0, stream>>>(hA2, attn_W3, attn_b3, atn, T_TOK, HID);

    // build g_i (f32 to d_out, padded bf16 mirror to ws)
    float* g = (float*)d_out;
    k_span_pool<<<N, 128, 0, stream>>>(states, embeds, atn, width_embed, starts, widths, g, gbf, N);

    // span score MLP (pure bf16 async-staged GEMMs)
    dim3 gS(HID / 128, (N + 127) / 128);
    k_gemm_bf16<<<gS, blk, 0, stream>>>(gbf, wS1t, score_b1, hS1, N, HID, GI_PAD, 1);
    k_gemm_bf16<<<gS, blk, 0, stream>>>(hS1, wS2t, score_b2, hS2, N, HID, HID, 1);
    k_matvec<<<(N + 7) / 8, 256, 0, stream>>>(hS2, score_W3, score_b3, g + (size_t)N * GI_D, N, HID);
}
